// LinearAttention_37237366456728
// MI455X (gfx1250) — matrix operand order repaired and checked by reading, not compiled, not run on hardware
//
#include <hip/hip_runtime.h>
#include <hip/hip_bf16.h>

// ---------------- problem constants (from reference) ----------------
#define HEADS   16
#define HD      64            // head dim
#define DMODEL  1024
#define BATCH   4
#define SEQ     2048
#define CHUNK   128
#define NCHUNK  (SEQ / CHUNK) // 16
#define MTOT    (BATCH * SEQ) // 8192

typedef __bf16 bf16;
typedef __attribute__((ext_vector_type(16))) __bf16 bf16x16;
typedef __attribute__((ext_vector_type(8)))  float  f32x8;
typedef __attribute__((ext_vector_type(4)))  unsigned int uint32x4;
typedef __attribute__((ext_vector_type(8)))  int    int32x8;
typedef __attribute__((ext_vector_type(4)))  int    int32x4;

__device__ __forceinline__ f32x8 wmma_bf16(bf16x16 a, bf16x16 b, f32x8 c) {
  // D = A(16x32) * B(32x16) + C, fp32 accumulate  ->  v_wmma_f32_16x16x32_bf16
  return __builtin_amdgcn_wmma_f32_16x16x32_bf16(false, a, false, b, (short)0, c,
                                                 false, false);
}

__device__ __forceinline__ f32x8 fzero() {
  f32x8 z;
#pragma unroll
  for (int i = 0; i < 8; ++i) z[i] = 0.f;
  return z;
}

// ---- WMMA fragment loaders (contiguous k = 16*(lane/16) + s, same for A and B) ----
__device__ __forceinline__ bf16x16 ldfrag_a(const bf16* p, int ld, int m0, int k0,
                                            int lane) {
  const bf16* row = p + (m0 + (lane & 15)) * ld + k0;
  int kg = (lane >> 4) << 4;
  bf16x16 a;
#pragma unroll
  for (int i = 0; i < 8; ++i) a[i] = row[kg + i];
#pragma unroll
  for (int i = 0; i < 8; ++i) a[8 + i] = row[8 + kg + i];
  return a;
}

// A = M^T, i.e. A[m,k] = Msrc[k,m], Msrc row-major stride ld.
__device__ __forceinline__ bf16x16 ldfrag_a_tr(const bf16* p, int ld, int m0, int k0,
                                               int lane) {
  int m = m0 + (lane & 15);
  int kg = k0 + ((lane >> 4) << 4);
  bf16x16 a;
#pragma unroll
  for (int i = 0; i < 8; ++i) a[i] = p[(kg + i) * ld + m];
#pragma unroll
  for (int i = 0; i < 8; ++i) a[8 + i] = p[(kg + 8 + i) * ld + m];
  return a;
}

// B fragment from row-major B (K x N), stride ld.
__device__ __forceinline__ bf16x16 ldfrag_b_rm(const bf16* p, int ld, int k0, int n0,
                                               int lane) {
  int n = n0 + (lane & 15);
  int kh = k0 + ((lane >> 4) << 4);
  bf16x16 b;
#pragma unroll
  for (int i = 0; i < 16; ++i) b[i] = p[(kh + i) * ld + n];
  return b;
}

// B = W^T: B[k,n] = W[n,k], W row-major stride ld (contiguous K reads).
__device__ __forceinline__ bf16x16 ldfrag_b_wt(const bf16* p, int ld, int k0, int n0,
                                               int lane) {
  const bf16* row = p + (n0 + (lane & 15)) * ld + k0 + ((lane >> 4) << 4);
  bf16x16 b;
#pragma unroll
  for (int i = 0; i < 16; ++i) b[i] = row[i];
  return b;
}

// ---- Tensor Data Mover: 2D bf16 tile (tile_dim0=64 x tile_dim1=rows) -> LDS ----
// D# packing per CDNA5 ISA ch.8 (group0 128b, group1 256b; groups 2/3 zero => 2D).
__device__ __forceinline__ void tdm_load_2d_bf16(const void* gptr,
                                                 unsigned int lds_byte, int rows,
                                                 int rows_total) {
  unsigned long long ga = (unsigned long long)gptr;
  uint32x4 g0;
  g0[0] = 1u;                                   // count=1, user mode, no gather
  g0[1] = lds_byte;                             // lds_addr
  g0[2] = (unsigned int)ga;                     // global_addr[31:0]
  g0[3] = (unsigned int)((ga >> 32) & 0x1FFFFFFu) | (2u << 30); // ga[56:32] | type=2
  int32x8 g1;
  g1[0] = (int)(1u << 16);                      // data_size=1 (2 bytes/elem)
  g1[1] = (int)(64u << 16);                     // tensor_dim0=64 (low16 @ bits 63:48)
  g1[2] = (int)(((unsigned int)rows_total & 0xFFFFu) << 16); // tensor_dim1 lo16
  g1[3] = (int)((((unsigned int)rows_total >> 16) & 0xFFFFu) |
                (64u << 16));                   // tensor_dim1 hi16 | tile_dim0=64
  g1[4] = rows & 0xFFFF;                        // tile_dim1=rows, tile_dim2=0
  g1[5] = 64;                                   // tensor_dim0_stride = 64 elems
  g1[6] = 0;
  g1[7] = 0;
  int32x4 z4 = {0, 0, 0, 0};
#if defined(__clang_major__) && (__clang_major__ >= 23)
  int32x8 z8 = {0, 0, 0, 0, 0, 0, 0, 0};
  __builtin_amdgcn_tensor_load_to_lds(g0, g1, z4, z4, z8, 0);
#else
  __builtin_amdgcn_tensor_load_to_lds(g0, g1, z4, z4, 0);
#endif
}

// ------------------------------------------------------------------
// GEMM:  C[m,n] = act( sum_k A[m,k] * W[n,k] + bias[n] )
// block tile 128(M) x 64(N), 8 waves, wave w owns rows [16w, 16w+16).
// ------------------------------------------------------------------
template <bool IN_BF16, bool SILU, bool HEAD_OUT>
__global__ void __launch_bounds__(256)
gemm_wmma(const void* __restrict__ Ain, const float* __restrict__ W,
          const float* __restrict__ bias, void* __restrict__ Out) {
  constexpr int K = DMODEL;
  __shared__ bf16 sA[128 * 32];
  __shared__ bf16 sB[64 * 32];

  const int tid  = threadIdx.x;
  const int lane = tid & 31;
  const int wave = tid >> 5;
  const int m0   = blockIdx.x * 128;
  const int n0   = blockIdx.y * 64;

  f32x8 acc[4];
#pragma unroll
  for (int t = 0; t < 4; ++t) acc[t] = fzero();

  const int r  = tid >> 1;          // 0..127  (row within A tile)
  const int c0 = (tid & 1) * 16;    // k-half

  for (int kt = 0; kt < K; kt += 32) {
    __syncthreads();
    if (IN_BF16) {
      const bf16* Ab = (const bf16*)Ain + (size_t)(m0 + r) * K + kt + c0;
#pragma unroll
      for (int i = 0; i < 16; ++i) sA[r * 32 + c0 + i] = Ab[i];
    } else {
      const float* Af = (const float*)Ain + (size_t)(m0 + r) * K + kt + c0;
#pragma unroll
      for (int i = 0; i < 16; ++i) sA[r * 32 + c0 + i] = (bf16)Af[i];
    }
    if (tid < 128) {
      const float* Wf = W + (size_t)(n0 + (tid >> 1)) * K + kt + c0;
#pragma unroll
      for (int i = 0; i < 16; ++i) sB[(tid >> 1) * 32 + c0 + i] = (bf16)Wf[i];
    }
    __syncthreads();

    if (kt + 32 < K) {  // gfx1250 global_prefetch_b8 on next tile
      __builtin_prefetch((const char*)Ain +
                             ((size_t)(m0 + r) * K + kt + 32 + c0) * (IN_BF16 ? 2 : 4),
                         0, 1);
    }

    bf16x16 a = ldfrag_a(sA, 32, wave * 16, 0, lane);
#pragma unroll
    for (int nt = 0; nt < 4; ++nt) {
      bf16x16 b = ldfrag_b_wt(sB, 32, 0, nt * 16, lane);
      acc[nt] = wmma_bf16(a, b, acc[nt]);
    }
  }

  const int mbase = m0 + wave * 16 + ((lane >> 4) << 3);
#pragma unroll
  for (int nt = 0; nt < 4; ++nt) {
    int n = n0 + nt * 16 + (lane & 15);
    float bia = bias[n];
#pragma unroll
    for (int i = 0; i < 8; ++i) {
      int m = mbase + i;
      float v = acc[nt][i] + bia;
      if (SILU) v = v / (1.f + __expf(-v));
      if (HEAD_OUT) {  // (B,L,D) -> (B,H,L,hd) bf16
        int bb = m / SEQ, l = m % SEQ;
        int h = n / HD, dd = n % HD;
        ((bf16*)Out)[(((size_t)(bb * HEADS + h)) * SEQ + l) * HD + dd] = (bf16)v;
      } else {
        ((float*)Out)[(size_t)m * DMODEL + n] = v;
      }
    }
  }
}

// ------------------------------------------------------------------
// Chunked causal linear attention for one (b,h): sequential over chunks.
// TDM double-buffers q/k/v chunk tiles; den_i = rowsum(masked scores) + q_i.z
// ------------------------------------------------------------------
#define QKV_BUF (3 * 128 * 64)  // bf16 elems per buffer set
#define ATT_SMEM (2 * QKV_BUF * 2 /*qkv dbl-buf*/ + 128 * 128 * 2 /*scores*/ + \
                  64 * 64 * 4 /*S f32*/ + 64 * 64 * 2 /*S bf16*/ + 64 * 4 + 128 * 4)

__global__ void __launch_bounds__(256)
linattn_wmma(const bf16* __restrict__ Q, const bf16* __restrict__ Km,
             const bf16* __restrict__ V, bf16* __restrict__ Outp) {
  extern __shared__ __align__(16) char smem[];
  bf16*  qkv = (bf16*)smem;                       // [2][3][128*64]
  bf16*  sc  = qkv + 2 * QKV_BUF;                 // 128x128 masked scores
  float* S   = (float*)(sc + 128 * 128);          // 64x64 running state (f32)
  bf16*  Sb  = (bf16*)(S + 64 * 64);              // bf16 snapshot of S
  float* zv  = (float*)(Sb + 64 * 64);            // 64
  float* den = zv + 64;                           // 128

  const int tid  = threadIdx.x;
  const int lane = tid & 31;
  const int wave = tid >> 5;
  const int bh   = blockIdx.x;           // b*H + h
  const int bat  = bh / HEADS;
  const int h    = bh % HEADS;
  const size_t base = (size_t)bh * SEQ * HD;

  for (int i = tid; i < 64 * 64; i += 256) S[i] = 0.f;
  if (tid < 64) zv[tid] = 0.f;

  // issue TDM loads for chunk 0 into buffer 0 (wave 0 only; EXEC-independent DMA)
  if (wave == 0) {
    unsigned int lb = (unsigned int)(unsigned long long)qkv;
    tdm_load_2d_bf16(Q + base, lb, CHUNK, SEQ);
    tdm_load_2d_bf16(Km + base, lb + 128 * 64 * 2, CHUNK, SEQ);
    tdm_load_2d_bf16(V + base, lb + 2 * 128 * 64 * 2, CHUNK, SEQ);
  }

  for (int nc = 0; nc < NCHUNK; ++nc) {
    bf16* qs = qkv + (nc & 1) * QKV_BUF;
    bf16* ks = qs + 128 * 64;
    bf16* vs = ks + 128 * 64;

    // snapshot S -> bf16 (exclusive prefix state used by this chunk)
    for (int i = tid; i < 64 * 64; i += 256) Sb[i] = (bf16)S[i];
    if (wave == 0) __builtin_amdgcn_s_wait_tensorcnt(0);  // chunk nc DMA done
    __syncthreads();  // qkv[nc&1] + Sb visible to all waves

    // overlap: kick off DMA of chunk nc+1 into the other buffer
    if (wave == 0 && nc + 1 < NCHUNK) {
      unsigned int lb = (unsigned int)(unsigned long long)(qkv + ((nc + 1) & 1) * QKV_BUF);
      const size_t go = base + (size_t)(nc + 1) * CHUNK * HD;
      int rt = SEQ - (nc + 1) * CHUNK;
      tdm_load_2d_bf16(Q + go, lb, CHUNK, rt);
      tdm_load_2d_bf16(Km + go, lb + 128 * 64 * 2, CHUNK, rt);
      tdm_load_2d_bf16(V + go, lb + 2 * 128 * 64 * 2, CHUNK, rt);
    }

    // ---- scores = q k^T (128x128, K=64), causal mask, store bf16 ----
    {
      int sm0 = wave * 16;
#pragma unroll 1
      for (int ntile = 0; ntile < 8; ++ntile) {
        f32x8 cacc = fzero();
#pragma unroll
        for (int kk = 0; kk < 64; kk += 32) {
          bf16x16 a = ldfrag_a(qs, HD, sm0, kk, lane);
          bf16x16 b = ldfrag_b_wt(ks, HD, kk, ntile * 16, lane);  // B[k,j]=k[j,k]
          cacc = wmma_bf16(a, b, cacc);
        }
        int j  = ntile * 16 + (lane & 15);
        int mb = sm0 + ((lane >> 4) << 3);
#pragma unroll
        for (int i = 0; i < 8; ++i) {
          int ii = mb + i;
          float s = (j <= ii) ? cacc[i] : 0.f;   // tril mask (incl. diagonal)
          sc[ii * 128 + j] = (bf16)s;
        }
      }
    }
    __syncthreads();

    // ---- den_i = rowsum(sc_i) + q_i . z ----
    if (tid < 128) {
      float s = 0.f;
      for (int j = 0; j < 128; ++j) s += (float)sc[tid * 128 + j];
      float qz = 0.f;
      for (int d_ = 0; d_ < HD; ++d_) qz += (float)qs[tid * HD + d_] * zv[d_];
      den[tid] = fmaxf(s + qz, 0.001f);
    }
    __syncthreads();

    // ---- num = sc @ v + q @ Sb ; write out ----
    {
      int sm0 = wave * 16;
#pragma unroll 1
      for (int nt = 0; nt < 4; ++nt) {
        f32x8 cacc = fzero();
#pragma unroll
        for (int kk = 0; kk < 128; kk += 32) {           // sc(128x128) @ v(128x64)
          bf16x16 a = ldfrag_a(sc, 128, sm0, kk, lane);
          bf16x16 b = ldfrag_b_rm(vs, HD, kk, nt * 16, lane);
          cacc = wmma_bf16(a, b, cacc);
        }
#pragma unroll
        for (int kk = 0; kk < 64; kk += 32) {            // q(128x64) @ S(64x64)
          bf16x16 a = ldfrag_a(qs, HD, sm0, kk, lane);
          bf16x16 b = ldfrag_b_rm(Sb, HD, kk, nt * 16, lane);
          cacc = wmma_bf16(a, b, cacc);
        }
        int e  = nt * 16 + (lane & 15);
        int mb = sm0 + ((lane >> 4) << 3);
#pragma unroll
        for (int i = 0; i < 8; ++i) {
          int row = mb + i;
          int l   = nc * CHUNK + row;
          float o = cacc[i] / den[row];
          Outp[((size_t)(bat * SEQ + l)) * DMODEL + h * HD + e] = (bf16)o;
        }
      }
    }

    // ---- S += k^T v (16 tiles 16x16, K=128; 2 tiles/wave); z += colsum(k) ----
    {
#pragma unroll
      for (int t = 0; t < 2; ++t) {
        int tile = wave * 2 + t;
        int dm0  = (tile >> 2) * 16;   // dd-tile
        int en0  = (tile & 3) * 16;    // e-tile
        int ncol = en0 + (lane & 15);
        int mb   = dm0 + ((lane >> 4) << 3);
        f32x8 cacc;
#pragma unroll
        for (int i = 0; i < 8; ++i) cacc[i] = S[(mb + i) * HD + ncol];
#pragma unroll
        for (int kk = 0; kk < 128; kk += 32) {
          bf16x16 a = ldfrag_a_tr(ks, HD, dm0, kk, lane);  // A[dd,j] = k[j,dd]
          bf16x16 b = ldfrag_b_rm(vs, HD, kk, en0, lane);
          cacc = wmma_bf16(a, b, cacc);
        }
#pragma unroll
        for (int i = 0; i < 8; ++i) S[(mb + i) * HD + ncol] = cacc[i];
      }
      if (tid < 64) {
        float s = zv[tid];
        for (int j = 0; j < 128; ++j) s += (float)ks[j * HD + tid];
        zv[tid] = s;
      }
    }
    __syncthreads();
  }
}

// ------------------------------------------------------------------
extern "C" void kernel_launch(void* const* d_in, const int* in_sizes, int n_in,
                              void* d_out, int out_size, void* d_ws, size_t ws_size,
                              hipStream_t stream) {
  const float* x  = (const float*)d_in[0];
  const float* Wq = (const float*)d_in[1];
  const float* bq = (const float*)d_in[2];
  const float* Wk = (const float*)d_in[3];
  const float* bk = (const float*)d_in[4];
  const float* Wv = (const float*)d_in[5];
  const float* bv = (const float*)d_in[6];
  const float* Wo = (const float*)d_in[7];
  const float* bo = (const float*)d_in[8];
  float* out = (float*)d_out;

  // workspace: q,k,v bf16 (B,H,L,hd) + attn bf16 (B,L,D)  = 64 MB
  bf16* qb = (bf16*)d_ws;
  bf16* kb = qb + (size_t)MTOT * DMODEL;
  bf16* vb = kb + (size_t)MTOT * DMODEL;
  bf16* ab = vb + (size_t)MTOT * DMODEL;

  dim3 grid(MTOT / 128, DMODEL / 64);
  gemm_wmma<false, true,  true ><<<grid, 256, 0, stream>>>(x,  Wq, bq, qb);
  gemm_wmma<false, true,  true ><<<grid, 256, 0, stream>>>(x,  Wk, bk, kb);
  gemm_wmma<false, false, true ><<<grid, 256, 0, stream>>>(x,  Wv, bv, vb);
  linattn_wmma<<<BATCH * HEADS, 256, ATT_SMEM, stream>>>(qb, kb, vb, ab);
  gemm_wmma<true,  false, false><<<grid, 256, 0, stream>>>(ab, Wo, bo, out);
}
